// MossAudioTokenizerMultiheadAttention_46849503264978
// MI455X (gfx1250) — matrix operand order repaired and checked by reading, not compiled, not run on hardware
//
#include <hip/hip_runtime.h>

// ---------------------------------------------------------------------------
// MossAudioTokenizer streaming MHA for gfx1250 (MI455X), wave32 + WMMA f16.
// B=16 T=8 H=16 D=64 C=1024 CAP=4096 CONTEXT=4096
// HBM-bound on the 536MB K/V stream -> staging loads are fully coalesced
// (contiguous 512B per load instruction); all matmuls via v_wmma_f32_16x16x32_f16.
// ---------------------------------------------------------------------------

typedef __attribute__((ext_vector_type(16))) _Float16 v16h;
typedef __attribute__((ext_vector_type(8)))  float    v8f;
typedef __attribute__((ext_vector_type(4)))  float    v4f;
typedef __attribute__((ext_vector_type(4)))  _Float16 v4h;

union Frag16 { v16h v; _Float16 e[16]; };

#define WMMA_F16(A, Bm, Cm) \
  __builtin_amdgcn_wmma_f32_16x16x32_f16(false, (A), false, (Bm), (short)0, (Cm), false, false)

__device__ __forceinline__ void wait_ds0() {
  asm volatile("s_wait_dscnt 0" ::: "memory");
}

// ---------------------------------------------------------------------------
// Kernel 1: qkv = x @ in_proj_w^T  (M=128, N=3072, K=1024), inline RoPE on q,k.
// One wave computes one 16x16 tile. Output f16 q/k/v in [B,H,T,D] workspace.
// (x and in_proj_w fit in L2; re-reads are cheap.)
// ---------------------------------------------------------------------------
__global__ __launch_bounds__(32) void k_qkv_rope(
    const float* __restrict__ x, const float* __restrict__ w,
    const int* __restrict__ end_offset,
    _Float16* __restrict__ wq, _Float16* __restrict__ wk, _Float16* __restrict__ wv)
{
  const int tile = blockIdx.x;
  const int tm = tile & 7;          // 8 m-tiles (128 rows)
  const int tn = tile >> 3;         // 192 n-tiles (3072 cols)
  const int lane = threadIdx.x;
  const int hi = lane >> 4, ln = lane & 15;
  const int m0 = tm * 16, n0 = tn * 16;

  const float* arow = x + (size_t)(m0 + ln) * 1024;            // A row m = ln
  const float* brow = w + (size_t)(n0 + ln) * 1024;            // B col n = ln

  v8f acc = {};
  for (int k0 = 0; k0 < 1024; k0 += 32) {
    Frag16 a, bm;
    // A 16x32 f16 layout: lane m=ln, k = hi*8 + (h&7) + (h>>3)*16
    const float* pa = arow + k0 + hi * 8;
    #pragma unroll
    for (int hh = 0; hh < 8; ++hh) { a.e[hh] = (_Float16)pa[hh]; a.e[8 + hh] = (_Float16)pa[16 + hh]; }
    // B 32x16 f16 layout: lane n=ln, k = hi*8 + (h&7) + (h>>3)*16
    const float* pb = brow + k0 + hi * 8;
    #pragma unroll
    for (int hh = 0; hh < 8; ++hh) { bm.e[hh] = (_Float16)pb[hh]; bm.e[8 + hh] = (_Float16)pb[16 + hh]; }
    acc = WMMA_F16(a.v, bm.v, acc);
  }

  // C layout: lane holds col n0+ln; vgpr r -> row m0 + r + hi*8  => b=2*tm+hi, t=r
  const int n_g = n0 + ln;
  const int sec = n_g >> 10;        // 0=q 1=k 2=v  (uniform per tile)
  const int c   = n_g & 1023;
  const int hH  = c >> 6;           // head (uniform per tile)
  const int d   = c & 63;
  const float sgn  = (d & 1) ? 1.f : -1.f;                       // even: -sin, odd: +sin
  const float freq = __expf((float)(d >> 1) * -0.28782313662425572f); // -ln(1e4)*2/64
  const int bI = 2 * tm + hi;
  const int eo = end_offset[bI];
  _Float16* dst = (sec == 0) ? wq : (sec == 1) ? wk : wv;

  #pragma unroll
  for (int r = 0; r < 8; ++r) {
    float val = acc[r];
    float partner = __shfl_xor(val, 1, 32);    // RoPE pair lives in adjacent lane
    if (sec < 2) {
      float sn, cs;
      __sincosf((float)(eo + r) * freq, &sn, &cs);
      val = val * cs + partner * (sgn * sn);
    }
    dst[((((size_t)bI * 16 + hH) * 8 + r) * 64) + d] = (_Float16)val;
  }
}

// ---------------------------------------------------------------------------
// Kernel 2: flash attention over the ring cache. One WG per (b,h), 4 waves.
// Caches are read-only; the T=8 freshly written slots are patched from ws.
// ---------------------------------------------------------------------------
__global__ __launch_bounds__(128) void k_attn(
    const float* __restrict__ cache_k, const float* __restrict__ cache_v,
    const int* __restrict__ end_offset,
    const _Float16* __restrict__ wq, const _Float16* __restrict__ wk,
    const _Float16* __restrict__ wv, _Float16* __restrict__ wo)
{
  __shared__ _Float16 qA[16][72];         // q rows, rows 8..15 zero
  __shared__ _Float16 Kb[4][32][72];      // per-wave K chunk [slot][d]
  __shared__ _Float16 Vt[4][64][40];      // per-wave V^T chunk [d][slot]
  __shared__ _Float16 Pb[4][16][40];      // per-wave P [t][slot]
  __shared__ float    mw[4][16];
  __shared__ float    lw[4][16];
  __shared__ float    Ob[4][16][64];

  const int b = blockIdx.x >> 4;
  const int h = blockIdx.x & 15;
  const int tid  = threadIdx.x;
  const int wave = tid >> 5;
  const int lane = tid & 31;
  const int hi = lane >> 4, ln = lane & 15;

  const int end     = end_offset[b];
  const int last    = end + 7;
  const int new_end = end + 8;
  const int e_idx   = last & 4095;
  const int wbase   = end & 4095;         // first ring slot overwritten this step

  for (int i = tid; i < 16 * 72; i += 128) (&qA[0][0])[i] = (_Float16)0.f;
  __syncthreads();
  const _Float16* qsrc = wq + (size_t)(b * 16 + h) * 8 * 64;
  for (int i = tid; i < 512; i += 128) qA[i >> 6][i & 63] = qsrc[i];
  __syncthreads();

  Frag16 qf[2];
  #pragma unroll
  for (int ks = 0; ks < 2; ++ks) {
    const _Float16* p = &qA[ln][ks * 32 + hi * 8];
    #pragma unroll
    for (int hh = 0; hh < 8; ++hh) { qf[ks].e[hh] = p[hh]; qf[ks].e[8 + hh] = p[16 + hh]; }
  }

  float mrow[8], lrow[8];
  #pragma unroll
  for (int r = 0; r < 8; ++r) { mrow[r] = -3.0e38f; lrow[r] = 0.f; }
  v8f o[4] = {{}, {}, {}, {}};

  const float* ckb = cache_k + (size_t)(b * 16 + h) * 4096 * 64;
  const float* cvb = cache_v + (size_t)(b * 16 + h) * 4096 * 64;
  const _Float16* wkb = wk + (size_t)(b * 16 + h) * 8 * 64;
  const _Float16* wvb = wv + (size_t)(b * 16 + h) * 8 * 64;

  for (int cix = wave; cix < 128; cix += 4) {   // 128 chunks of 32 slots
    const int s0 = cix * 32;

    // ---- stage K chunk + V^T chunk, fully coalesced ----
    // 32 slots x 64 f32 = 512 float4's; lane i of instruction j loads
    // float4 #(j*32+i): every load instruction spans 512 contiguous bytes.
    #pragma unroll
    for (int i = 0; i < 16; ++i) {
      const int idx = i * 32 + lane;       // 0..511
      const int sl  = idx >> 4;            // slot within chunk 0..31
      const int d0  = (idx & 15) * 4;      // dim group
      const int slot = s0 + sl;
      int tw = slot - wbase; if (tw < 0) tw += 4096;
      v4h kh, vh;
      if (tw < 8) {                        // ring-write patch: fresh RoPE'd k / new v
        kh = *(const v4h*)(wkb + tw * 64 + d0);
        vh = *(const v4h*)(wvb + tw * 64 + d0);
      } else {
        v4f kv = *(const v4f*)(ckb + (size_t)slot * 64 + d0);
        v4f vv = *(const v4f*)(cvb + (size_t)slot * 64 + d0);
        #pragma unroll
        for (int j = 0; j < 4; ++j) { kh[j] = (_Float16)kv[j]; vh[j] = (_Float16)vv[j]; }
      }
      *(v4h*)&Kb[wave][sl][d0] = kh;
      #pragma unroll
      for (int j = 0; j < 4; ++j) Vt[wave][d0 + j][sl] = vh[j];
    }
    wait_ds0();   // wave-private buffers: DS-counter wait, no block barrier needed

    // ---- scores: S[t, s] = q·k / 8 with streaming ring mask ----
    float sv[2][8], pvv[2][8], cmax[8];
    #pragma unroll
    for (int r = 0; r < 8; ++r) cmax[r] = -3.0e38f;

    #pragma unroll
    for (int sub = 0; sub < 2; ++sub) {
      v8f sa = {};
      #pragma unroll
      for (int ks = 0; ks < 2; ++ks) {
        Frag16 kb;
        const _Float16* p = &Kb[wave][sub * 16 + ln][ks * 32 + hi * 8];
        #pragma unroll
        for (int hh = 0; hh < 8; ++hh) { kb.e[hh] = p[hh]; kb.e[8 + hh] = p[16 + hh]; }
        sa = WMMA_F16(qf[ks].v, kb.v, sa);
      }
      const int slot  = s0 + sub * 16 + ln;
      const int delta = slot - e_idx;
      const int pos   = (delta <= 0) ? (last + delta) : (last + delta - 4096);
      const bool valid0 = slot < new_end;
      #pragma unroll
      for (int r = 0; r < 8; ++r) {
        const int t  = r + hi * 8;          // rows 8..15 are padding (junk, unstored)
        const int dd = end + t - pos;
        const bool ok = valid0 && (dd >= 0) && (dd < 4096);
        float s = ok ? sa[r] * 0.125f : -3.0e38f;
        sv[sub][r] = s;
        cmax[r] = fmaxf(cmax[r], s);
      }
    }

    // ---- online softmax update (row reductions over each 16-lane half) ----
    #pragma unroll
    for (int r = 0; r < 8; ++r) {
      float cm = cmax[r];
      #pragma unroll
      for (int msk = 1; msk <= 8; msk <<= 1) cm = fmaxf(cm, __shfl_xor(cm, msk, 32));
      const float mnew  = fmaxf(mrow[r], cm);
      const float alpha = __expf(mrow[r] - mnew);   // finite sentinel: never NaN
      mrow[r] = mnew;
      float p0 = __expf(sv[0][r] - mnew);
      float p1 = __expf(sv[1][r] - mnew);
      pvv[0][r] = p0; pvv[1][r] = p1;
      float rs = p0 + p1;
      #pragma unroll
      for (int msk = 1; msk <= 8; msk <<= 1) rs += __shfl_xor(rs, msk, 32);
      lrow[r] = lrow[r] * alpha + rs;
      #pragma unroll
      for (int dt = 0; dt < 4; ++dt) o[dt][r] *= alpha;
    }

    // ---- P -> LDS (relayout C->A), then O += P @ V ----
    #pragma unroll
    for (int sub = 0; sub < 2; ++sub) {
      #pragma unroll
      for (int r = 0; r < 8; ++r)
        Pb[wave][r + hi * 8][sub * 16 + ln] = (_Float16)pvv[sub][r];
    }
    wait_ds0();

    Frag16 pa;
    {
      const _Float16* pp = &Pb[wave][ln][hi * 8];
      #pragma unroll
      for (int hh = 0; hh < 8; ++hh) { pa.e[hh] = pp[hh]; pa.e[8 + hh] = pp[16 + hh]; }
    }
    #pragma unroll
    for (int dt = 0; dt < 4; ++dt) {
      Frag16 vb;
      const _Float16* pv2 = &Vt[wave][dt * 16 + ln][hi * 8];
      #pragma unroll
      for (int hh = 0; hh < 8; ++hh) { vb.e[hh] = pv2[hh]; vb.e[8 + hh] = pv2[16 + hh]; }
      o[dt] = WMMA_F16(pa.v, vb.v, o[dt]);
    }
  }

  // ---- merge the 4 waves' partial softmax states ----
  if (ln == 0) {
    #pragma unroll
    for (int r = 0; r < 8; ++r) { mw[wave][r + hi * 8] = mrow[r]; lw[wave][r + hi * 8] = lrow[r]; }
  }
  #pragma unroll
  for (int dt = 0; dt < 4; ++dt) {
    #pragma unroll
    for (int r = 0; r < 8; ++r)
      Ob[wave][r + hi * 8][dt * 16 + ln] = o[dt][r];
  }
  __syncthreads();

  for (int i = tid; i < 512; i += 128) {
    const int t = i >> 6, d = i & 63;
    const float a0 = mw[0][t], a1 = mw[1][t], a2 = mw[2][t], a3 = mw[3][t];
    const float mt = fmaxf(fmaxf(a0, a1), fmaxf(a2, a3));
    const float s0 = __expf(a0 - mt), s1 = __expf(a1 - mt);
    const float s2 = __expf(a2 - mt), s3 = __expf(a3 - mt);
    const float lt = s0 * lw[0][t] + s1 * lw[1][t] + s2 * lw[2][t] + s3 * lw[3][t];
    const float ov = s0 * Ob[0][t][d] + s1 * Ob[1][t][d] + s2 * Ob[2][t][d] + s3 * Ob[3][t][d];
    wo[(((size_t)b * 8 + t) * 16 + h) * 64 + d] = (_Float16)(ov / lt);   // [B,T,H,D]=[B,T,C]
  }
}

// ---------------------------------------------------------------------------
// Kernel 3: out = attn @ out_proj_w^T  (M=128, N=1024, K=1024), fp32 store.
// ---------------------------------------------------------------------------
__global__ __launch_bounds__(32) void k_outproj(
    const _Float16* __restrict__ attn, const float* __restrict__ w,
    float* __restrict__ out)
{
  const int tile = blockIdx.x;
  const int tm = tile & 7;          // 8 m-tiles
  const int tn = tile >> 3;         // 64 n-tiles
  const int lane = threadIdx.x;
  const int hi = lane >> 4, ln = lane & 15;
  const int m0 = tm * 16, n0 = tn * 16;

  const _Float16* arow = attn + (size_t)(m0 + ln) * 1024;
  const float*    brow = w    + (size_t)(n0 + ln) * 1024;

  v8f acc = {};
  for (int k0 = 0; k0 < 1024; k0 += 32) {
    Frag16 a, bm;
    const _Float16* pa = arow + k0 + hi * 8;
    #pragma unroll
    for (int hh = 0; hh < 8; ++hh) { a.e[hh] = pa[hh]; a.e[8 + hh] = pa[16 + hh]; }
    const float* pb = brow + k0 + hi * 8;
    #pragma unroll
    for (int hh = 0; hh < 8; ++hh) { bm.e[hh] = (_Float16)pb[hh]; bm.e[8 + hh] = (_Float16)pb[16 + hh]; }
    acc = WMMA_F16(a.v, bm.v, acc);
  }
  #pragma unroll
  for (int r = 0; r < 8; ++r)
    out[(size_t)(m0 + r + hi * 8) * 1024 + n0 + ln] = acc[r];
}

// ---------------------------------------------------------------------------
extern "C" void kernel_launch(void* const* d_in, const int* in_sizes, int n_in,
                              void* d_out, int out_size, void* d_ws, size_t ws_size,
                              hipStream_t stream) {
  (void)in_sizes; (void)n_in; (void)out_size; (void)ws_size;
  const float* x       = (const float*)d_in[0];
  const float* in_w    = (const float*)d_in[1];
  const float* out_w   = (const float*)d_in[2];
  const float* cache_k = (const float*)d_in[3];
  const float* cache_v = (const float*)d_in[4];
  const int*   end_off = (const int*)d_in[5];
  float* out = (float*)d_out;

  // Workspace: q/k/v (RoPE'd, f16, [B,H,T,D]) + attention output (f16, [B,T,C]).
  _Float16* wq = (_Float16*)d_ws;
  _Float16* wk = wq + 16 * 16 * 8 * 64;
  _Float16* wv = wk + 16 * 16 * 8 * 64;
  _Float16* wo = wv + 16 * 16 * 8 * 64;     // 1 MB total

  k_qkv_rope<<<dim3(8 * 192), dim3(32), 0, stream>>>(x, in_w, end_off, wq, wk, wv);
  k_attn   <<<dim3(16 * 16),  dim3(128), 0, stream>>>(cache_k, cache_v, end_off, wq, wk, wv, wo);
  k_outproj<<<dim3(8 * 64),   dim3(32), 0, stream>>>(wo, out_w, out);
}
